// GraphEdgeAttenNetwork_13924283974460
// MI455X (gfx1250) — matrix operand order repaired and checked by reading, not compiled, not run on hardware
//
#include <hip/hip_runtime.h>

typedef __attribute__((ext_vector_type(16))) _Float16 v16h;
typedef __attribute__((ext_vector_type(8)))  _Float16 v8h;
typedef __attribute__((ext_vector_type(8)))  float    v8f;

static __device__ inline v16h cat8(v8h lo, v8h hi) {
  return __builtin_shufflevector(lo, hi, 0, 1, 2, 3, 4, 5, 6, 7,
                                 8, 9, 10, 11, 12, 13, 14, 15);
}

static __device__ inline v8f wmma(v16h a, v16h b, v8f c) {
  return __builtin_amdgcn_wmma_f32_16x16x32_f16(false, a, false, b, (short)0, c,
                                                false, false);
}

// A fragment: 16xK tile (row-major, leading dim ld) in LDS -> two ds_load_b128.
static __device__ inline v16h frag_A_lds(const _Float16* base, int ld, int lane) {
  int m = lane & 15, g = (lane >> 4) & 1;
  const _Float16* p = base + m * ld + 8 * g;
  return cat8(*(const v8h*)p, *(const v8h*)(p + 16));
}

// B fragment from pre-packed weights (one contiguous 32B load per lane).
static __device__ inline v16h frag_B_pk(const _Float16* __restrict__ Wp, int tile, int lane) {
  const _Float16* p = Wp + ((size_t)tile << 9) + lane * 16;
  return cat8(*(const v8h*)p, *(const v8h*)(p + 8));
}

static __device__ inline v8f frag_C_bias(const float* __restrict__ bias, int n0, int lane) {
  float bv = bias[n0 + (lane & 15)];
  v8f c;
#pragma unroll
  for (int r = 0; r < 8; ++r) c[r] = bv;
  return c;
}

// monotone float <-> signed-int key (for atomicMax-based segment max)
static __device__ inline int f2key(float f) {
  int b = __float_as_int(f);
  return b >= 0 ? b : (int)0x80000000 - b;
}
static __device__ inline float key2f(int k) {
  int b = k >= 0 ? k : (int)0x80000000 - k;
  return __int_as_float(b);
}

// ---------- utility kernels ----------

__global__ void f32_to_f16_kernel(const float* __restrict__ s, _Float16* __restrict__ d, int n) {
  int i = blockIdx.x * 256 + threadIdx.x;
  if (i < n) d[i] = (_Float16)s[i];
}

// Pack row-major fp32 weight W[K][Nc] into WMMA-B fragment order (f16):
//   Wp[((kk*(Nc/16)+nt)*32 + lane)*16 + t] = W[kk*32 + 16*(t>>3) + 8*(lane>>4) + (t&7)][nt*16 + (lane&15)]
__global__ void pack_B_kernel(const float* __restrict__ W, _Float16* __restrict__ Wp,
                              int K, int Nc) {
  int idx = blockIdx.x * 256 + threadIdx.x;
  if (idx >= K * Nc) return;
  int t = idx & 15;
  int lane = (idx >> 4) & 31;
  int tile = idx >> 9;
  int tilesN = Nc >> 4;
  int nt = tile % tilesN, kk = tile / tilesN;
  int k = kk * 32 + ((t >> 3) << 4) + ((lane >> 4) << 3) + (t & 7);
  int col = nt * 16 + (lane & 15);
  Wp[idx] = (_Float16)W[(size_t)k * Nc + col];
}

__global__ void init_agg_kernel(int* __restrict__ agg, int n) {
  int i = blockIdx.x * 256 + threadIdx.x;
  if (i < n) agg[i] = f2key(-INFINITY);
}

// ---------- edge kernel: one wave = 32 edges (two 16-row A tiles share B) ----
// LDS carve (57344 B, single wave per block):
//   0     XI  32x128 f16      (later: ATT 32x32 f16)
//   8192  EF  32x128 f16  \   (later: LOG 32x8x16 f32, spans 8192..24576)
//   16384 XJ  32x128 f16  /
//   24576 HID 32x256 f16      (later: VAL 32x128 f32)
//   40960 Q   head-major [h][e(32)][c(16)] f16
//   49152 EP  head-major [h][e(32)][c(16)] f16

__global__ __launch_bounds__(32) void edge_kernel(
    const float* __restrict__ edge_feature, const int* __restrict__ eidx, int E,
    const _Float16* __restrict__ x16,
    const _Float16* __restrict__ We1, const float* __restrict__ be1,
    const _Float16* __restrict__ We2, const float* __restrict__ be2,
    const _Float16* __restrict__ Wqh, const float* __restrict__ bq,
    const _Float16* __restrict__ Weh, const float* __restrict__ beb,
    const _Float16* __restrict__ Wvh, const float* __restrict__ bv,
    const _Float16* __restrict__ Wa1, const float* __restrict__ ba1,
    const _Float16* __restrict__ Wa2, const float* __restrict__ ba2,
    float* __restrict__ gcn_out, int* __restrict__ aggBits) {
  __shared__ __align__(16) char sm[57344];
  const int lane = threadIdx.x & 31;
  _Float16* sXI  = (_Float16*)(sm + 0);
  _Float16* sEF  = (_Float16*)(sm + 8192);
  _Float16* sXJ  = (_Float16*)(sm + 16384);
  _Float16* sHID = (_Float16*)(sm + 24576);
  float*    sVAL = (float*)(sm + 24576);
  _Float16* sQ   = (_Float16*)(sm + 40960);
  _Float16* sEP  = (_Float16*)(sm + 49152);
  _Float16* sATT = (_Float16*)(sm + 0);
  float*    sLOG = (float*)(sm + 8192);

  long ntiles = (E + 31) / 32;
  long tile = blockIdx.x;
  if (tile >= ntiles) tile = ntiles - 1;
  long e0l = tile * 32;
  if (e0l > (long)E - 32) e0l = (long)E - 32;
  const int e0 = (int)e0l;

  const int* __restrict__ srcI = eidx;
  const int* __restrict__ dstI = eidx + E;

  // ---- gather xi/xj (pre-converted f16) and convert edge_feature tile ----
  {
    const int c0 = lane * 4;
#pragma unroll 4
    for (int r = 0; r < 32; ++r) {
      int e = e0 + r;
      int si = srcI[e], dj = dstI[e];
      *(uint2*)&sXI[r * 128 + c0] = *(const uint2*)&x16[(size_t)si * 128 + c0];
      *(uint2*)&sXJ[r * 128 + c0] = *(const uint2*)&x16[(size_t)dj * 128 + c0];
      float4 v = *(const float4*)&edge_feature[(size_t)e * 128 + c0];
      sEF[r * 128 + c0 + 0] = (_Float16)v.x;
      sEF[r * 128 + c0 + 1] = (_Float16)v.y;
      sEF[r * 128 + c0 + 2] = (_Float16)v.z;
      sEF[r * 128 + c0 + 3] = (_Float16)v.w;
    }
  }
  __syncthreads();

  const int n = lane & 15, g = (lane >> 4) & 1;

  // ---- GEMM1: hid = relu([xi|ef|xj] @ We1 + be1)   [32x384]@[384x256] ----
  for (int nt = 0; nt < 16; ++nt) {
    v8f c0 = frag_C_bias(be1, nt * 16, lane);
    v8f c1 = c0;
#pragma unroll
    for (int kk = 0; kk < 12; ++kk) {
      int k0 = kk * 32;
      const _Float16* srcm = (k0 < 128) ? sXI : (k0 < 256 ? sEF : sXJ);
      const _Float16* pa = srcm + (k0 & 127);
      v16h b = frag_B_pk(We1, kk * 16 + nt, lane);
      c0 = wmma(frag_A_lds(pa, 128, lane), b, c0);
      c1 = wmma(frag_A_lds(pa + 16 * 128, 128, lane), b, c1);
    }
#pragma unroll
    for (int r = 0; r < 8; ++r) {
      int m = r + 8 * g;
      float v0 = c0[r]; if (v0 < 0.f) v0 = 0.f;
      float v1 = c1[r]; if (v1 < 0.f) v1 = 0.f;
      sHID[m * 256 + nt * 16 + n]        = (_Float16)v0;
      sHID[(m + 16) * 256 + nt * 16 + n] = (_Float16)v1;
    }
  }
  __syncthreads();

  // ---- GEMM2: gcn_edge = hid @ We2 + be2  -> global f32 ----
  for (int nt = 0; nt < 8; ++nt) {
    v8f c0 = frag_C_bias(be2, nt * 16, lane);
    v8f c1 = c0;
#pragma unroll
    for (int kk = 0; kk < 8; ++kk) {
      v16h b = frag_B_pk(We2, kk * 8 + nt, lane);
      c0 = wmma(frag_A_lds(sHID + kk * 32, 256, lane), b, c0);
      c1 = wmma(frag_A_lds(sHID + 16 * 256 + kk * 32, 256, lane), b, c1);
    }
#pragma unroll
    for (int r = 0; r < 8; ++r) {
      int m = r + 8 * g;
      gcn_out[(size_t)(e0 + m) * 128 + nt * 16 + n]      = c0[r];
      gcn_out[(size_t)(e0 + 16 + m) * 128 + nt * 16 + n] = c1[r];
    }
  }
  __syncthreads();

  // ---- GEMM3/4/5: q = xi@Wq, ep = ef@We, val = xj@Wv (val overwrites HID) ----
  // q/ep stored head-major: s[(h*32 + e)*16 + c] = proj[e][c*8 + h]
  for (int nt = 0; nt < 8; ++nt) {
    v8f cq0 = frag_C_bias(bq,  nt * 16, lane), cq1 = cq0;
    v8f ce0 = frag_C_bias(beb, nt * 16, lane), ce1 = ce0;
    v8f cv0 = frag_C_bias(bv,  nt * 16, lane), cv1 = cv0;
#pragma unroll
    for (int kk = 0; kk < 4; ++kk) {
      v16h bqf = frag_B_pk(Wqh, kk * 8 + nt, lane);
      v16h bef = frag_B_pk(Weh, kk * 8 + nt, lane);
      v16h bvf = frag_B_pk(Wvh, kk * 8 + nt, lane);
      cq0 = wmma(frag_A_lds(sXI + kk * 32, 128, lane), bqf, cq0);
      cq1 = wmma(frag_A_lds(sXI + 2048 + kk * 32, 128, lane), bqf, cq1);
      ce0 = wmma(frag_A_lds(sEF + kk * 32, 128, lane), bef, ce0);
      ce1 = wmma(frag_A_lds(sEF + 2048 + kk * 32, 128, lane), bef, ce1);
      cv0 = wmma(frag_A_lds(sXJ + kk * 32, 128, lane), bvf, cv0);
      cv1 = wmma(frag_A_lds(sXJ + 2048 + kk * 32, 128, lane), bvf, cv1);
    }
    int col = nt * 16 + n;
    int hh = col & 7, cc = col >> 3;
#pragma unroll
    for (int r = 0; r < 8; ++r) {
      int m = r + 8 * g;
      sQ[(hh * 32 + m) * 16 + cc]        = (_Float16)cq0[r];
      sQ[(hh * 32 + m + 16) * 16 + cc]   = (_Float16)cq1[r];
      sEP[(hh * 32 + m) * 16 + cc]       = (_Float16)ce0[r];
      sEP[(hh * 32 + m + 16) * 16 + cc]  = (_Float16)ce1[r];
      sVAL[m * 128 + col]        = cv0[r];
      sVAL[(m + 16) * 128 + col] = cv1[r];
    }
  }
  __syncthreads();

  // ---- attention MLP per head: [32x32]@[32x32] relu -> [32x32]@[32x16] ----
  for (int h = 0; h < 8; ++h) {
    int m = lane & 15;
    const _Float16* pq = sQ  + (h * 32 + m) * 16 + 8 * g;
    const _Float16* pe = sEP + (h * 32 + m) * 16 + 8 * g;
    v16h a0 = cat8(*(const v8h*)pq, *(const v8h*)pe);
    v16h a1 = cat8(*(const v8h*)(pq + 256), *(const v8h*)(pe + 256));
#pragma unroll
    for (int nt = 0; nt < 2; ++nt) {
      v16h b = frag_B_pk(Wa1, nt, lane);
      v8f c0 = frag_C_bias(ba1, nt * 16, lane);
      v8f c1 = c0;
      c0 = wmma(a0, b, c0);
      c1 = wmma(a1, b, c1);
#pragma unroll
      for (int r = 0; r < 8; ++r) {
        int mm = r + 8 * g;
        float v0 = c0[r]; if (v0 < 0.f) v0 = 0.f;
        float v1 = c1[r]; if (v1 < 0.f) v1 = 0.f;
        sATT[mm * 32 + nt * 16 + n]        = (_Float16)v0;
        sATT[(mm + 16) * 32 + nt * 16 + n] = (_Float16)v1;
      }
    }
    __syncthreads();
    {
      v16h b2 = frag_B_pk(Wa2, 0, lane);
      v8f cl0 = frag_C_bias(ba2, 0, lane);
      v8f cl1 = cl0;
      cl0 = wmma(frag_A_lds(sATT, 32, lane), b2, cl0);
      cl1 = wmma(frag_A_lds(sATT + 16 * 32, 32, lane), b2, cl1);
#pragma unroll
      for (int r = 0; r < 8; ++r) {
        int mm = r + 8 * g;
        sLOG[(mm * 8 + h) * 16 + n]        = cl0[r];  // [e][h][o]
        sLOG[((mm + 16) * 8 + h) * 16 + n] = cl1[r];
      }
    }
    __syncthreads();
  }

  // ---- softmax over o per (e,h); xx = prob*value; scatter-max per src node ----
#pragma unroll
  for (int p = 0; p < 8; ++p) {
    int pair = lane + 32 * p;          // 256 (e,h) pairs
    int el = pair >> 3, h = pair & 7;
    float lg[16], mx = -INFINITY;
#pragma unroll
    for (int o = 0; o < 16; ++o) { lg[o] = sLOG[(el * 8 + h) * 16 + o]; mx = fmaxf(mx, lg[o]); }
    float s = 0.f;
#pragma unroll
    for (int o = 0; o < 16; ++o) { lg[o] = __expf(lg[o] - mx); s += lg[o]; }
    float inv = 1.f / s;
    int node = srcI[e0 + el];
#pragma unroll
    for (int o = 0; o < 16; ++o) {
      float xx = lg[o] * inv * sVAL[el * 128 + o * 8 + h];
      atomicMax(&aggBits[(size_t)node * 128 + o * 8 + h], f2key(xx));
    }
  }
}

// ---------- node kernel: out = relu([x|agg]@Wp1+bp1)@Wp2+bp2, 32 nodes/wave ----
__global__ __launch_bounds__(32) void node_kernel(
    const _Float16* __restrict__ x16, const int* __restrict__ aggBits,
    const _Float16* __restrict__ Wp1, const float* __restrict__ bp1,
    const _Float16* __restrict__ Wp2, const float* __restrict__ bp2,
    float* __restrict__ out, int N) {
  __shared__ __align__(16) char sm[32768];
  const int lane = threadIdx.x & 31;
  _Float16* sH   = (_Float16*)sm;             // [32][256] = [x | agg]
  _Float16* sHID = (_Float16*)(sm + 16384);   // [32][256]

  long ntiles = (N + 31) / 32;
  long tile = blockIdx.x;
  if (tile >= ntiles) tile = ntiles - 1;
  long n0l = tile * 32;
  if (n0l > (long)N - 32) n0l = (long)N - 32;
  const int n0i = (int)n0l;

  {
    const int c0 = lane * 4;
#pragma unroll 4
    for (int r = 0; r < 32; ++r) {
      int node = n0i + r;
      *(uint2*)&sH[r * 256 + c0] = *(const uint2*)&x16[(size_t)node * 128 + c0];
#pragma unroll
      for (int u = 0; u < 4; ++u) {
        float f = key2f(aggBits[(size_t)node * 128 + c0 + u]);
        if (f == -INFINITY) f = 0.f;       // empty segment -> 0 (torch_scatter)
        sH[r * 256 + 128 + c0 + u] = (_Float16)f;
      }
    }
  }
  __syncthreads();

  const int n = lane & 15, g = (lane >> 4) & 1;
  for (int nt = 0; nt < 16; ++nt) {
    v8f c0 = frag_C_bias(bp1, nt * 16, lane);
    v8f c1 = c0;
#pragma unroll
    for (int kk = 0; kk < 8; ++kk) {
      v16h b = frag_B_pk(Wp1, kk * 16 + nt, lane);
      c0 = wmma(frag_A_lds(sH + kk * 32, 256, lane), b, c0);
      c1 = wmma(frag_A_lds(sH + 16 * 256 + kk * 32, 256, lane), b, c1);
    }
#pragma unroll
    for (int r = 0; r < 8; ++r) {
      int m = r + 8 * g;
      float v0 = c0[r]; if (v0 < 0.f) v0 = 0.f;
      float v1 = c1[r]; if (v1 < 0.f) v1 = 0.f;
      sHID[m * 256 + nt * 16 + n]        = (_Float16)v0;
      sHID[(m + 16) * 256 + nt * 16 + n] = (_Float16)v1;
    }
  }
  __syncthreads();
  for (int nt = 0; nt < 8; ++nt) {
    v8f c0 = frag_C_bias(bp2, nt * 16, lane);
    v8f c1 = c0;
#pragma unroll
    for (int kk = 0; kk < 8; ++kk) {
      v16h b = frag_B_pk(Wp2, kk * 8 + nt, lane);
      c0 = wmma(frag_A_lds(sHID + kk * 32, 256, lane), b, c0);
      c1 = wmma(frag_A_lds(sHID + 16 * 256 + kk * 32, 256, lane), b, c1);
    }
#pragma unroll
    for (int r = 0; r < 8; ++r) {
      int m = r + 8 * g;
      out[(size_t)(n0i + m) * 128 + nt * 16 + n]      = c0[r];
      out[(size_t)(n0i + 16 + m) * 128 + nt * 16 + n] = c1[r];
    }
  }
}

// ---------- host ----------
extern "C" void kernel_launch(void* const* d_in, const int* in_sizes, int n_in,
                              void* d_out, int out_size, void* d_ws, size_t ws_size,
                              hipStream_t stream) {
  const float* x    = (const float*)d_in[0];
  const float* ef   = (const float*)d_in[1];
  const int*   eidx = (const int*)d_in[2];
  const float* W_e1 = (const float*)d_in[3];  const float* b_e1 = (const float*)d_in[4];
  const float* W_e2 = (const float*)d_in[5];  const float* b_e2 = (const float*)d_in[6];
  const float* Wq   = (const float*)d_in[7];  const float* bq   = (const float*)d_in[8];
  const float* We   = (const float*)d_in[9];  const float* be   = (const float*)d_in[10];
  const float* Wv   = (const float*)d_in[11]; const float* bv   = (const float*)d_in[12];
  const float* Wa1  = (const float*)d_in[13]; const float* ba1  = (const float*)d_in[14];
  const float* Wa2  = (const float*)d_in[15]; const float* ba2  = (const float*)d_in[16];
  const float* Wp1  = (const float*)d_in[17]; const float* bp1  = (const float*)d_in[18];
  const float* Wp2  = (const float*)d_in[19]; const float* bp2  = (const float*)d_in[20];

  const int N = in_sizes[0] / 128;
  const int E = in_sizes[1] / 128;

  char* ws = (char*)d_ws;
  size_t off = 0;
  auto carve = [&](size_t bytes) -> char* {
    char* p = ws + off;
    off += (bytes + 255) & ~(size_t)255;
    return p;
  };
  _Float16* x16   = (_Float16*)carve((size_t)N * 128 * 2);
  _Float16* We1h  = (_Float16*)carve((size_t)in_sizes[3] * 2);
  _Float16* We2h  = (_Float16*)carve((size_t)in_sizes[5] * 2);
  _Float16* Wqh   = (_Float16*)carve((size_t)in_sizes[7] * 2);
  _Float16* Weh   = (_Float16*)carve((size_t)in_sizes[9] * 2);
  _Float16* Wvh   = (_Float16*)carve((size_t)in_sizes[11] * 2);
  _Float16* Wa1h  = (_Float16*)carve((size_t)in_sizes[13] * 2);
  _Float16* Wa2h  = (_Float16*)carve((size_t)in_sizes[15] * 2);
  _Float16* Wp1h  = (_Float16*)carve((size_t)in_sizes[17] * 2);
  _Float16* Wp2h  = (_Float16*)carve((size_t)in_sizes[19] * 2);
  int* aggBits    = (int*)carve((size_t)N * 128 * 4);

  f32_to_f16_kernel<<<(N * 128 + 255) / 256, 256, 0, stream>>>(x, x16, N * 128);

  auto pack = [&](const float* s, _Float16* d, int K, int Nc) {
    pack_B_kernel<<<(K * Nc + 255) / 256, 256, 0, stream>>>(s, d, K, Nc);
  };
  pack(W_e1, We1h, 384, 256);
  pack(W_e2, We2h, 256, 128);
  pack(Wq,   Wqh,  128, 128);
  pack(We,   Weh,  128, 128);
  pack(Wv,   Wvh,  128, 128);
  pack(Wa1,  Wa1h, 32, 32);
  pack(Wa2,  Wa2h, 32, 16);
  pack(Wp1,  Wp1h, 256, 256);
  pack(Wp2,  Wp2h, 256, 128);

  init_agg_kernel<<<(N * 128 + 255) / 256, 256, 0, stream>>>(aggBits, N * 128);

  float* out_ptr = (float*)d_out;             // [N,128] first
  float* gcn_ptr = out_ptr + (size_t)N * 128; // then [E,128]

  const int ntE = (E + 31) / 32;
  edge_kernel<<<ntE, 32, 0, stream>>>(
      ef, eidx, E, x16, We1h, b_e1, We2h, b_e2, Wqh, bq, Weh, be, Wvh, bv,
      Wa1h, ba1, Wa2h, ba2, gcn_ptr, aggBits);

  const int ntN = (N + 31) / 32;
  node_kernel<<<ntN, 32, 0, stream>>>(
      x16, aggBits, Wp1h, bp1, Wp2h, bp2, out_ptr, N);
}
